// RecurrentClassifier_54107997995192
// MI455X (gfx1250) — matrix operand order repaired and checked by reading, not compiled, not run on hardware
//
#include <hip/hip_runtime.h>
#include <hip/hip_bf16.h>

typedef __attribute__((ext_vector_type(16))) _Float16 v16h;
typedef __attribute__((ext_vector_type(8)))  _Float16 v8h;
typedef __attribute__((ext_vector_type(8)))  float    v8f;

#define B_  128
#define I_  256
#define H_  512
#define NC_ 16
#define T_  24
#define M_  14
#define G4  2048   // 4*H

__device__ inline float sigmoidf_(float x){ return 1.f/(1.f+__expf(-x)); }
__device__ inline float tanhf_(float x){ return 2.f/(1.f+__expf(-2.f*x)) - 1.f; }

__device__ inline v8f wmma_f16(v16h a, v16h b, v8f c){
  return __builtin_amdgcn_wmma_f32_16x16x32_f16(false, a, false, b, (short)0, c, false, false);
}

// A fragment (16x32, f16) from a row-major [rows][ldk] f16 matrix.
// lanes 0-15 hold K=0..15, lanes 16-31 hold K=16..31 (matches B), M=lane%16.
__device__ inline v16h load_a_frag(const _Float16* __restrict__ base, int ldk,
                                   int row, int kbase, int lr, int half){
  const _Float16* p = base + (size_t)(row + lr)*ldk + kbase + half*16;
  return *(const v16h*)p;
}

// B fragment (32x16, f16): column n = W row n (W stored [n][k] row-major);
// lanes 0-15 hold K=0..15, lanes 16-31 hold K=16..31 -> contiguous 32B per lane.
__device__ inline v16h load_b_frag(const _Float16* __restrict__ base, int ldk,
                                   int n, int kbase, int half){
  return *(const v16h*)(base + (size_t)n*ldk + kbase + half*16);
}

// ---------------- prep: f32->f16 conversions, x transpose, state zeroing ----
__global__ __launch_bounds__(256) void prep_kernel(
    const float* __restrict__ x, const float* __restrict__ Wih, const float* __restrict__ Whh,
    _Float16* __restrict__ wih16, _Float16* __restrict__ whh16, _Float16* __restrict__ xbf,
    _Float16* __restrict__ h16, float* __restrict__ Cpack, float* __restrict__ hdot,
    float* __restrict__ Pout)
{
  const int n1 = G4*I_;        // 524288
  const int n2 = G4*H_;        // 1048576
  const int n3 = T_*B_*I_;     // 786432
  const int n4 = 2*B_*H_;      // 131072 (h ping-pong, both buffers)
  const int n5 = B_*H_;        // 65536  (cell state fragments)
  const int n6 = M_*B_;        // 1792   (halt logit partials)
  const int n7 = B_;           // 128    (P accumulator in d_out)
  const int total = n1+n2+n3+n4+n5+n6+n7;
  for (int i = blockIdx.x*blockDim.x + threadIdx.x; i < total; i += gridDim.x*blockDim.x) {
    int k = i;
    if (k < n1) { wih16[k] = (_Float16)Wih[k]; continue; } k -= n1;
    if (k < n2) { whh16[k] = (_Float16)Whh[k]; continue; } k -= n2;
    if (k < n3) {
      int t = k / (B_*I_); int r = k % (B_*I_); int b = r / I_; int ii = r % I_;
      xbf[k] = (_Float16)x[((size_t)b*I_ + ii)*T_ + t];
      continue; } k -= n3;
    if (k < n4) { h16[k] = (_Float16)0.f; continue; } k -= n4;
    if (k < n5) { Cpack[k] = 0.f; continue; } k -= n5;
    if (k < n6) { hdot[k] = 0.f; continue; } k -= n6;
    Pout[k] = 0.f;
  }
}

// ---------------- GX[t] = x_t @ W_ih^T + (b_ih + b_hh), stored in D-frag layout
__global__ __launch_bounds__(256) void gx_kernel(
    const _Float16* __restrict__ xbf, const _Float16* __restrict__ wih16,
    const float* __restrict__ b_ih, const float* __restrict__ b_hh,
    float* __restrict__ GX)
{
  int gid  = blockIdx.x*256 + threadIdx.x;
  int w    = gid >> 5;                 // wave id 0..24575
  int ln   = threadIdx.x & 31, half = ln >> 4, lr = ln & 15;
  int t    = w >> 10;                  // 1024 waves per timestep
  int rem  = w & 1023;
  int mt   = rem >> 7;                 // 8 row tiles
  int nt2  = rem & 127;                // 128 gate-column tiles
  int n    = nt2*16 + lr;              // gate row of W_ih (0..2047)
  const _Float16* xt = xbf + (size_t)t*B_*I_;
  v8f acc = {};
#pragma unroll
  for (int kt = 0; kt < 8; kt++) {
    v16h a = load_a_frag(xt, I_, mt*16, kt*32, lr, half);
    v16h b = load_b_frag(wih16, I_, n, kt*32, half);
    acc = wmma_f16(a, b, acc);
  }
  float bias = b_ih[n] + b_hh[n];
  int gate = n >> 9, jt = (n & 511) >> 4;
  float* out = GX + ((size_t)(((t*4 + gate)*8 + mt)*32 + jt)*32 + ln)*8;
#pragma unroll
  for (int v = 0; v < 8; v++) out[v] = acc[v] + bias;
}

// ---------------- one ACT tick: gates = GX + h@Whh^T, LSTM update, halt partials
__global__ __launch_bounds__(256) void tick_kernel(
    const _Float16* __restrict__ hin, _Float16* __restrict__ hout,
    const _Float16* __restrict__ whh16, const float* __restrict__ GX,
    float* __restrict__ Cpack, float* __restrict__ s_hist, float* __restrict__ c_hist,
    float* __restrict__ hdot, const float* __restrict__ halt_w, int t, int m)
{
  int gid = blockIdx.x*256 + threadIdx.x;
  int w   = gid >> 5;                  // 0..255  (8 mt x 32 jt)
  int ln  = threadIdx.x & 31, half = ln >> 4, lr = ln & 15;
  int mt  = w >> 5, jt = w & 31;

  v8f ai = {}, af = {}, ag = {}, ao = {};
  for (int kt = 0; kt < 16; kt++) {
    v16h a = load_a_frag(hin, H_, mt*16, kt*32, lr, half);
    v16h b0 = load_b_frag(whh16, H_, 0*H_ + jt*16 + lr, kt*32, half);
    ai = wmma_f16(a, b0, ai);
    v16h b1 = load_b_frag(whh16, H_, 1*H_ + jt*16 + lr, kt*32, half);
    af = wmma_f16(a, b1, af);
    v16h b2 = load_b_frag(whh16, H_, 2*H_ + jt*16 + lr, kt*32, half);
    ag = wmma_f16(a, b2, ag);
    v16h b3 = load_b_frag(whh16, H_, 3*H_ + jt*16 + lr, kt*32, half);
    ao = wmma_f16(a, b3, ao);
  }

  const float* gxb = GX + ((size_t)((t*4 + 0)*8 + mt)*32 + jt)*256 + ln*8;
  const size_t gstride = (size_t)8*32*256;  // 65536 floats per gate plane
  v8f gi = ai + *(const v8f*)(gxb + 0*gstride);
  v8f gf = af + *(const v8f*)(gxb + 1*gstride);
  v8f gg = ag + *(const v8f*)(gxb + 2*gstride);
  v8f go = ao + *(const v8f*)(gxb + 3*gstride);

  int e0 = ((mt*32 + jt)*32 + ln)*8;
  v8f cold = *(const v8f*)(Cpack + e0);
  int j = jt*16 + lr;
  float hw = halt_w[j];
  float hn[8];
#pragma unroll
  for (int v = 0; v < 8; v++) {
    float iv = sigmoidf_(gi[v]);
    float fv = sigmoidf_(gf[v]);
    float gv = tanhf_(gg[v]);
    float ov = sigmoidf_(go[v]);
    float cn = fv*cold[v] + iv*gv;
    float h  = ov*tanhf_(cn);
    Cpack[e0 + v] = cn;
    c_hist[(size_t)m*(B_*H_) + e0 + v] = cn;
    s_hist[(size_t)m*(B_*H_) + e0 + v] = h;
    hn[v] = h;
    int b = mt*16 + v + half*8;
    hout[(size_t)b*H_ + j] = (_Float16)h;
  }
  // halting-logit partial: reduce this wave's 16 hidden columns per row, accumulate
#pragma unroll
  for (int v = 0; v < 8; v++) {
    float s = hn[v]*hw;
    s += __shfl_xor(s, 1, 32);
    s += __shfl_xor(s, 2, 32);
    s += __shfl_xor(s, 4, 32);
    s += __shfl_xor(s, 8, 32);
    if (lr == 0) {
      int b = mt*16 + v + half*8;
      atomicAdd(&hdot[m*B_ + b], s);
    }
  }
}

// ---------------- per-timestep ACT combine: ph weights, st/ct, yt, pt, nt -----
__global__ __launch_bounds__(256) void combine_kernel(
    float* __restrict__ hdot, const float* __restrict__ halt_b,
    const float* __restrict__ s_hist, const float* __restrict__ c_hist,
    float* __restrict__ Cpack, _Float16* __restrict__ h16b0, float* __restrict__ st32,
    const float* __restrict__ dec_w, const float* __restrict__ dec_b,
    float* __restrict__ Yout, float* __restrict__ Pout, float* __restrict__ Nout, int t)
{
  __shared__ float phs[M_][B_];
  __shared__ unsigned andmask;
  int tid = threadIdx.x;
  if (tid == 0) andmask = 0x3FFFu;
  __syncthreads();

  float pc[M_];
  if (tid < B_) {
    float hb = halt_b[0];
    float cum = 0.f;
    unsigned mymask = 0;
    for (int m = 0; m < M_; m++) {
      float d = hdot[m*B_ + tid];
      hdot[m*B_ + tid] = 0.f;               // re-zero for next timestep's ticks
      cum += sigmoidf_(d + hb);
      pc[m] = cum;
      if (cum >= 1.f - 0.01f) mymask |= (1u << m);
    }
    atomicAnd(&andmask, mymask);
  }
  __syncthreads();
  unsigned am = andmask;
  if (tid < B_) {
    int ns = M_;
    for (int m = 0; m < M_; m++) if ((am >> m) & 1u) { ns = m + 1; break; }
    float pmax[M_];
    for (int m = 0; m < M_; m++) pmax[m] = (m >= ns - 1) ? 1.f : fminf(1.f, pc[m]);
    float prev = 0.f;
    int nt = M_ - 1; bool found = false;
    for (int m = 0; m < M_; m++) {
      phs[m][tid] = pmax[m] - prev;
      prev = pmax[m];
      if (!found && pmax[m] >= 1.f) { nt = m; found = true; }
    }
    float rt = 1.f - ((nt == 0) ? pmax[M_-1] : pmax[nt-1]);
    Pout[tid] += (float)nt + rt;
    Nout[tid*T_ + t] = (float)nt;
  }
  __syncthreads();

  // st/ct = sum_m ph[m,b] * hist[m]  (fragment layout)
  for (int e = tid; e < B_*H_; e += 256) {
    int v    = e & 7;
    int ln   = (e >> 3) & 31;
    int tile = e >> 8;
    int mt = tile >> 5, jt = tile & 31;
    int b = mt*16 + v + ((ln >> 4) << 3);
    int j = jt*16 + (ln & 15);
    float ss = 0.f, cc = 0.f;
    for (int m = 0; m < M_; m++) {
      float wgt = phs[m][b];
      ss += wgt * s_hist[(size_t)m*(B_*H_) + e];
      cc += wgt * c_hist[(size_t)m*(B_*H_) + e];
    }
    Cpack[e] = cc;
    st32[(size_t)b*H_ + j] = ss;
    h16b0[(size_t)b*H_ + j] = (_Float16)ss;   // next timestep reads buffer 0
  }
  __syncthreads();

  // yt = st @ dec_w^T + dec_b  (exact: sum_m ph = 1, y linear in h)
  for (int o = tid; o < B_*NC_; o += 256) {
    int b = o >> 4, nc = o & 15;
    float s = dec_b[nc];
    for (int j = 0; j < H_; j++) s += st32[(size_t)b*H_ + j] * dec_w[nc*H_ + j];
    Yout[((size_t)b*NC_ + nc)*T_ + t] = s;
  }
}

extern "C" void kernel_launch(void* const* d_in, const int* in_sizes, int n_in,
                              void* d_out, int out_size, void* d_ws, size_t ws_size,
                              hipStream_t stream)
{
  (void)in_sizes; (void)n_in; (void)out_size; (void)ws_size;
  const float* x      = (const float*)d_in[0];
  const float* Wih    = (const float*)d_in[1];
  const float* Whh    = (const float*)d_in[2];
  const float* b_ih   = (const float*)d_in[3];
  const float* b_hh   = (const float*)d_in[4];
  const float* halt_w = (const float*)d_in[5];
  const float* halt_b = (const float*)d_in[6];
  const float* dec_w  = (const float*)d_in[7];
  const float* dec_b  = (const float*)d_in[8];

  float* Yout = (float*)d_out;                 // [128,16,24]
  float* Pout = Yout + B_*NC_*T_;              // [128]
  float* Nout = Pout + B_;                     // [128,24] (int values stored as float)

  char* ws = (char*)d_ws;
  size_t off = 0;
  auto take = [&](size_t bytes) -> char* {
    char* p = ws + off; off = (off + bytes + 255) & ~(size_t)255; return p;
  };
  _Float16* wih16  = (_Float16*)take((size_t)G4*I_*2);       // 1 MB
  _Float16* whh16  = (_Float16*)take((size_t)G4*H_*2);       // 2 MB
  _Float16* xbf    = (_Float16*)take((size_t)T_*B_*I_*2);    // 1.5 MB
  float*    GX     = (float*)   take((size_t)T_*4*8*32*256*4); // 24 MB
  _Float16* h16    = (_Float16*)take((size_t)2*B_*H_*2);     // 256 KB ping-pong
  float*    Cpack  = (float*)   take((size_t)B_*H_*4);       // 256 KB
  float*    s_hist = (float*)   take((size_t)M_*B_*H_*4);    // 3.5 MB
  float*    c_hist = (float*)   take((size_t)M_*B_*H_*4);    // 3.5 MB
  float*    hdot   = (float*)   take((size_t)M_*B_*4);       // 7 KB
  float*    st32   = (float*)   take((size_t)B_*H_*4);       // 256 KB

  prep_kernel<<<2048, 256, 0, stream>>>(x, Wih, Whh, wih16, whh16, xbf,
                                        h16, Cpack, hdot, Pout);
  gx_kernel<<<3072, 256, 0, stream>>>(xbf, wih16, b_ih, b_hh, GX);

  for (int t = 0; t < T_; t++) {
    for (int m = 0; m < M_; m++) {
      const _Float16* hin = h16 + (size_t)(m & 1)*B_*H_;
      _Float16* hout      = h16 + (size_t)((m + 1) & 1)*B_*H_;
      tick_kernel<<<32, 256, 0, stream>>>(hin, hout, whh16, GX, Cpack,
                                          s_hist, c_hist, hdot, halt_w, t, m);
    }
    combine_kernel<<<1, 256, 0, stream>>>(hdot, halt_b, s_hist, c_hist, Cpack,
                                          h16 /*buffer 0*/, st32, dec_w, dec_b,
                                          Yout, Pout, Nout, t);
  }
}
